// AttentionBlock_88699664597518
// MI455X (gfx1250) — matrix operand order repaired and checked by reading, not compiled, not run on hardware
//
#include <hip/hip_runtime.h>
#include <stdint.h>

typedef unsigned short u16t;
typedef __attribute__((ext_vector_type(16))) __bf16 v16bf;
typedef __attribute__((ext_vector_type(8)))  float  v8f;

union Frag { v16bf h; uint4 u[2]; };

#define BATCH   2
#define CCH     256
#define NHEADS  8
#define HD      32
#define SEQ     4096
#define NGROUPS 32
#define GCH     8

__device__ __forceinline__ u16t f2bf(float f){
  union { float f; unsigned int u; } cv; cv.f = f;
  unsigned int u = cv.u;
  unsigned int r = (u + 0x7FFFu + ((u >> 16) & 1u)) >> 16;
  return (u16t)r;
}

// Async memory->LDS copy (CDNA5, tracked by ASYNCcnt). LDS operand is the
// wave-relative LDS byte offset = low 32 bits of the generic pointer.
__device__ __forceinline__ void async_b128(void* lds_ptr, const void* gptr){
  unsigned int lds = (unsigned int)(uintptr_t)lds_ptr;
  asm volatile("global_load_async_to_lds_b128 %0, %1, off"
               :: "v"(lds), "v"(gptr) : "memory");
}
__device__ __forceinline__ void wait_async0(){
  asm volatile("s_wait_asynccnt 0x0" ::: "memory");
}

// ---------------------------------------------------------------- weights->bf16
__global__ __launch_bounds__(256)
void cvt_weights_kernel(const float* qkvw, const float* projw,
                        u16t* qkvw_bf, u16t* projw_bf){
  int i = blockIdx.x * 256 + threadIdx.x;
  if (i < 3*CCH*CCH) qkvw_bf[i] = f2bf(qkvw[i]);
  if (i < CCH*CCH)   projw_bf[i] = f2bf(projw[i]);
}

// ---------------------------------------------------------------- GroupNorm -> xn_T[b][s][c] (bf16)
__global__ __launch_bounds__(256)
void groupnorm_kernel(const float* __restrict__ x, const float* __restrict__ gw,
                      const float* __restrict__ gb, u16t* __restrict__ xnT){
  __shared__ float red[256];
  const int tid = threadIdx.x;
  const int bg = blockIdx.x, b = bg >> 5, g = bg & 31;
  const float* xp = x + ((long)(b*CCH + g*GCH))*SEQ;
  float s1 = 0.f, s2 = 0.f;
  for (int i = tid; i < GCH*SEQ; i += 256){ float v = xp[i]; s1 += v; s2 += v*v; }
  red[tid] = s1; __syncthreads();
  for (int st = 128; st; st >>= 1){ if (tid < st) red[tid] += red[tid+st]; __syncthreads(); }
  const float mean = red[0] * (1.0f/(GCH*SEQ));
  __syncthreads();
  red[tid] = s2; __syncthreads();
  for (int st = 128; st; st >>= 1){ if (tid < st) red[tid] += red[tid+st]; __syncthreads(); }
  const float var  = red[0] * (1.0f/(GCH*SEQ)) - mean*mean;
  const float rstd = rsqrtf(var + 1e-5f);
  for (int i = tid; i < GCH*SEQ; i += 256){
    int ci = i >> 12, s = i & (SEQ-1);
    int c = g*GCH + ci;
    float v = (xp[i] - mean) * rstd * gw[c] + gb[c];
    xnT[((long)b*SEQ + s)*CCH + c] = f2bf(v);
  }
}

// ---------------------------------------------------------------- QKV GEMM (async double-buffered)
__global__ __launch_bounds__(256)
void qkv_gemm_kernel(const u16t* __restrict__ W, const u16t* __restrict__ X,
                     const float* __restrict__ bias,
                     u16t* __restrict__ qT, u16t* __restrict__ kT, u16t* __restrict__ v){
  __shared__ u16t lA[2][64][32];
  __shared__ u16t lB[2][128][32];
  const int tid = threadIdx.x, lane = tid & 31, wave = tid >> 5;
  const int wm = wave & 3, wn = wave >> 2;
  const int l16 = lane & 15, lh = lane >> 4;
  const int m0 = blockIdx.y * 64, n0 = blockIdx.x * 128, b = blockIdx.z;
  const u16t* Xb = X + (long)b * SEQ * CCH;
  const int am = tid >> 2, ach = tid & 3;      // A staging coords
  const int bn = tid >> 1, bhf = tid & 1;      // B staging coords

  auto stage = [&](int buf, int k0){
    async_b128(&lA[buf][am][ach*8], &W[(long)(m0+am)*CCH + k0 + ach*8]);
    const u16t* src = &Xb[(long)(n0+bn)*CCH + k0 + bhf*16];
    async_b128(&lB[buf][bn][bhf*16],     src);
    async_b128(&lB[buf][bn][bhf*16 + 8], src + 8);
  };

  v8f acc[4]; acc[0] = {}; acc[1] = {}; acc[2] = {}; acc[3] = {};
  stage(0, 0);
  for (int k0 = 0; k0 < CCH; k0 += 32){
    const int buf = (k0 >> 5) & 1;
    wait_async0();
    __syncthreads();
    if (k0 + 32 < CCH) stage(buf ^ 1, k0 + 32);
    Frag a;
    a.u[0] = *(const uint4*)&lA[buf][wm*16 + l16][lh*16];
    a.u[1] = *(const uint4*)&lA[buf][wm*16 + l16][lh*16 + 8];
    #pragma unroll
    for (int t = 0; t < 4; ++t){
      Frag bf;
      int n = wn*64 + t*16 + l16;
      bf.u[0] = *(const uint4*)&lB[buf][n][lh*16];
      bf.u[1] = *(const uint4*)&lB[buf][n][lh*16 + 8];
      acc[t] = __builtin_amdgcn_wmma_f32_16x16x32_bf16(false, a.h, false, bf.h,
                                                       (short)0, acc[t], false, false);
    }
  }
  const float scale = 0.17677669529663687f; // 1/sqrt(32)
  #pragma unroll
  for (int t = 0; t < 4; ++t){
    int s = n0 + wn*64 + t*16 + l16;
    #pragma unroll
    for (int r = 0; r < 8; ++r){
      int o = m0 + wm*16 + lh*8 + r;
      float val = acc[t][r] + bias[o];
      int which = o >> 8, rem = o & 255, head = rem >> 5, d = rem & 31;
      long base = (long)(b*NHEADS + head);
      if (which == 0)      qT[(base*SEQ + s)*HD + d] = f2bf(val * scale);
      else if (which == 1) kT[(base*SEQ + s)*HD + d] = f2bf(val);
      else                 v [(base*HD + d)*SEQ + s] = f2bf(val);
    }
  }
}

// ---------------------------------------------------------------- Flash attention (64-key tiles)
__global__ __launch_bounds__(256)
void flash_attn_kernel(const u16t* __restrict__ qT, const u16t* __restrict__ kT,
                       const u16t* __restrict__ v, u16t* __restrict__ attnT){
  __shared__ u16t lP[8][16][64];
  const int tid = threadIdx.x, lane = tid & 31, wave = tid >> 5;
  const int l16 = lane & 15, lh = lane >> 4;
  const int bh = blockIdx.y, b = bh >> 3, head = bh & 7;
  const int sq = blockIdx.x * 128 + wave * 16;

  const u16t* qrow = qT + ((long)bh*SEQ + sq + l16)*HD;
  Frag qf;
  qf.u[0] = *(const uint4*)&qrow[lh*16];
  qf.u[1] = *(const uint4*)&qrow[lh*16 + 8];

  v8f acc0 = {}, acc1 = {};
  const v8f zero = {};
  float rm[8], rl[8];
  #pragma unroll
  for (int r = 0; r < 8; ++r){ rm[r] = -1e30f; rl[r] = 0.f; }

  for (int t0 = 0; t0 < SEQ; t0 += 64){
    // ---- S = q . k^T for 64 keys: 4 WMMAs ----
    const u16t* krow = kT + ((long)bh*SEQ + t0 + l16)*HD + lh*16;
    Frag kb[4];
    #pragma unroll
    for (int j = 0; j < 4; ++j){
      kb[j].u[0] = *(const uint4*)&krow[j*16*HD];
      kb[j].u[1] = *(const uint4*)&krow[j*16*HD + 8];
    }
    v8f sS[4];
    #pragma unroll
    for (int j = 0; j < 4; ++j)
      sS[j] = __builtin_amdgcn_wmma_f32_16x16x32_bf16(false, qf.h, false, kb[j].h,
                                                      (short)0, zero, false, false);
    if (t0 + 64 < SEQ){
      __builtin_prefetch(krow + 64*HD, 0, 1);                           // next K tile
      __builtin_prefetch(v + ((long)bh*HD + l16)*SEQ + t0 + 64, 0, 1);  // next V tile
    }

    // ---- online softmax over the 64-key tile ----
    #pragma unroll
    for (int r = 0; r < 8; ++r){
      float a0 = sS[0][r], a1 = sS[1][r], a2 = sS[2][r], a3 = sS[3][r];
      float mx = fmaxf(fmaxf(a0, a1), fmaxf(a2, a3));
      mx = fmaxf(mx, __shfl_xor(mx, 1, 32));
      mx = fmaxf(mx, __shfl_xor(mx, 2, 32));
      mx = fmaxf(mx, __shfl_xor(mx, 4, 32));
      mx = fmaxf(mx, __shfl_xor(mx, 8, 32));
      float nm    = fmaxf(rm[r], mx);
      float alpha = __expf(rm[r] - nm);
      float p0 = __expf(a0 - nm), p1 = __expf(a1 - nm);
      float p2 = __expf(a2 - nm), p3 = __expf(a3 - nm);
      float ps = (p0 + p1) + (p2 + p3);
      ps += __shfl_xor(ps, 1, 32);
      ps += __shfl_xor(ps, 2, 32);
      ps += __shfl_xor(ps, 4, 32);
      ps += __shfl_xor(ps, 8, 32);
      rl[r] = rl[r]*alpha + ps;
      rm[r] = nm;
      acc0[r] *= alpha; acc1[r] *= alpha;
      int m = r + lh*8;                    // C-layout row
      lP[wave][m][l16]      = f2bf(p0);
      lP[wave][m][16 + l16] = f2bf(p1);
      lP[wave][m][32 + l16] = f2bf(p2);
      lP[wave][m][48 + l16] = f2bf(p3);
    }
    asm volatile("s_wait_dscnt 0x0" ::: "memory"); // wave-local LDS fence

    // ---- P (A-layout) and V (B-layout) fragments, 4 PV WMMAs ----
    Frag pf0, pf1;
    pf0.u[0] = *(const uint4*)&lP[wave][l16][lh*16];
    pf0.u[1] = *(const uint4*)&lP[wave][l16][lh*16 + 8];
    pf1.u[0] = *(const uint4*)&lP[wave][l16][32 + lh*16];
    pf1.u[1] = *(const uint4*)&lP[wave][l16][32 + lh*16 + 8];

    const u16t* vrow0 = v + ((long)bh*HD + l16)*SEQ + t0 + lh*16;  // d 0..15
    const u16t* vrow1 = vrow0 + 16*SEQ;                            // d 16..31
    Frag vb00, vb01, vb10, vb11;                                   // [dhalf][thalf]
    vb00.u[0] = *(const uint4*)&vrow0[0];  vb00.u[1] = *(const uint4*)&vrow0[8];
    vb01.u[0] = *(const uint4*)&vrow0[32]; vb01.u[1] = *(const uint4*)&vrow0[40];
    vb10.u[0] = *(const uint4*)&vrow1[0];  vb10.u[1] = *(const uint4*)&vrow1[8];
    vb11.u[0] = *(const uint4*)&vrow1[32]; vb11.u[1] = *(const uint4*)&vrow1[40];

    acc0 = __builtin_amdgcn_wmma_f32_16x16x32_bf16(false, pf0.h, false, vb00.h,
                                                   (short)0, acc0, false, false);
    acc0 = __builtin_amdgcn_wmma_f32_16x16x32_bf16(false, pf1.h, false, vb01.h,
                                                   (short)0, acc0, false, false);
    acc1 = __builtin_amdgcn_wmma_f32_16x16x32_bf16(false, pf0.h, false, vb10.h,
                                                   (short)0, acc1, false, false);
    acc1 = __builtin_amdgcn_wmma_f32_16x16x32_bf16(false, pf1.h, false, vb11.h,
                                                   (short)0, acc1, false, false);
  }

  #pragma unroll
  for (int r = 0; r < 8; ++r){
    float inv = 1.0f / rl[r];
    int s = sq + r + lh*8;
    long orow = ((long)b*SEQ + s)*CCH + head*HD;
    attnT[orow + l16]      = f2bf(acc0[r]*inv);
    attnT[orow + 16 + l16] = f2bf(acc1[r]*inv);
  }
}

// ---------------------------------------------------------------- Proj GEMM + bias + residual (f32 out)
__global__ __launch_bounds__(256)
void proj_gemm_kernel(const u16t* __restrict__ W, const u16t* __restrict__ X,
                      const float* __restrict__ bias, const float* __restrict__ xin,
                      float* __restrict__ out){
  __shared__ u16t lA[2][64][32];
  __shared__ u16t lB[2][128][32];
  const int tid = threadIdx.x, lane = tid & 31, wave = tid >> 5;
  const int wm = wave & 3, wn = wave >> 2;
  const int l16 = lane & 15, lh = lane >> 4;
  const int m0 = blockIdx.y * 64, n0 = blockIdx.x * 128, b = blockIdx.z;
  const u16t* Xb = X + (long)b * SEQ * CCH;
  const int am = tid >> 2, ach = tid & 3;
  const int bn = tid >> 1, bhf = tid & 1;

  auto stage = [&](int buf, int k0){
    async_b128(&lA[buf][am][ach*8], &W[(long)(m0+am)*CCH + k0 + ach*8]);
    const u16t* src = &Xb[(long)(n0+bn)*CCH + k0 + bhf*16];
    async_b128(&lB[buf][bn][bhf*16],     src);
    async_b128(&lB[buf][bn][bhf*16 + 8], src + 8);
  };

  v8f acc[4]; acc[0] = {}; acc[1] = {}; acc[2] = {}; acc[3] = {};
  stage(0, 0);
  for (int k0 = 0; k0 < CCH; k0 += 32){
    const int buf = (k0 >> 5) & 1;
    wait_async0();
    __syncthreads();
    if (k0 + 32 < CCH) stage(buf ^ 1, k0 + 32);
    Frag a;
    a.u[0] = *(const uint4*)&lA[buf][wm*16 + l16][lh*16];
    a.u[1] = *(const uint4*)&lA[buf][wm*16 + l16][lh*16 + 8];
    #pragma unroll
    for (int t = 0; t < 4; ++t){
      Frag bf;
      int n = wn*64 + t*16 + l16;
      bf.u[0] = *(const uint4*)&lB[buf][n][lh*16];
      bf.u[1] = *(const uint4*)&lB[buf][n][lh*16 + 8];
      acc[t] = __builtin_amdgcn_wmma_f32_16x16x32_bf16(false, a.h, false, bf.h,
                                                       (short)0, acc[t], false, false);
    }
  }
  #pragma unroll
  for (int t = 0; t < 4; ++t){
    int s = n0 + wn*64 + t*16 + l16;
    #pragma unroll
    for (int r = 0; r < 8; ++r){
      int o = m0 + wm*16 + lh*8 + r;
      long idx = ((long)(b*CCH + o))*SEQ + s;
      out[idx] = acc[t][r] + bias[o] + xin[idx];
    }
  }
}

// ---------------------------------------------------------------- launch
extern "C" void kernel_launch(void* const* d_in, const int* in_sizes, int n_in,
                              void* d_out, int out_size, void* d_ws, size_t ws_size,
                              hipStream_t stream){
  const float* x      = (const float*)d_in[0];
  const float* gn_w   = (const float*)d_in[1];
  const float* gn_b   = (const float*)d_in[2];
  const float* qkv_w  = (const float*)d_in[3];
  const float* qkv_b  = (const float*)d_in[4];
  const float* proj_w = (const float*)d_in[5];
  const float* proj_b = (const float*)d_in[6];
  float* out = (float*)d_out;

  char* ws = (char*)d_ws;
  size_t off = 0;
  const size_t actBytes  = (size_t)BATCH*SEQ*CCH*2;            // 4 MB each
  u16t* xnT   = (u16t*)(ws + off); off += actBytes;
  u16t* qT    = (u16t*)(ws + off); off += actBytes;
  u16t* kT    = (u16t*)(ws + off); off += actBytes;
  u16t* vB    = (u16t*)(ws + off); off += actBytes;
  u16t* attnT = (u16t*)(ws + off); off += actBytes;
  u16t* qkvw_bf  = (u16t*)(ws + off); off += (size_t)3*CCH*CCH*2;
  u16t* projw_bf = (u16t*)(ws + off); off += (size_t)CCH*CCH*2;

  cvt_weights_kernel<<<dim3((3*CCH*CCH + 255)/256), 256, 0, stream>>>(
      qkv_w, proj_w, qkvw_bf, projw_bf);
  groupnorm_kernel<<<dim3(BATCH*NGROUPS), 256, 0, stream>>>(x, gn_w, gn_b, xnT);
  qkv_gemm_kernel<<<dim3(SEQ/128, (3*CCH)/64, BATCH), 256, 0, stream>>>(
      qkvw_bf, xnT, qkv_b, qT, kT, vB);
  flash_attn_kernel<<<dim3(SEQ/128, BATCH*NHEADS), 256, 0, stream>>>(qT, kT, vB, attnT);
  proj_gemm_kernel<<<dim3(SEQ/128, CCH/64, BATCH), 256, 0, stream>>>(
      projw_bf, attnT, proj_b, x, out);
}
